// _SepsisGAT_38723425140784
// MI455X (gfx1250) — matrix operand order repaired and checked by reading, not compiled, not run on hardware
//
#include <hip/hip_runtime.h>
#include <hip/hip_bf16.h>

typedef __attribute__((ext_vector_type(16))) _Float16 v16h;
typedef __attribute__((ext_vector_type(8)))  float    v8f;

#define NEGSLOPE 0.2f
#define BNEPS    1e-5f

// ---------------------------------------------------------------------------
// helpers: monotone float <-> uint encoding for atomicMax on floats
// ---------------------------------------------------------------------------
__device__ __forceinline__ unsigned fenc(float f) {
  unsigned u = __float_as_uint(f);
  return (u & 0x80000000u) ? ~u : (u | 0x80000000u);
}
__device__ __forceinline__ float fdec(unsigned u) {
  u = (u & 0x80000000u) ? (u & 0x7fffffffu) : ~u;
  return __uint_as_float(u);
}

// ---------------------------------------------------------------------------
// zero fill (u32 pattern; works for f32 0.0 and encoded -inf init)
// ---------------------------------------------------------------------------
__global__ void gat_zero(unsigned* __restrict__ p, long long n) {
  long long i = (long long)blockIdx.x * blockDim.x + threadIdx.x;
  if (i < n) p[i] = 0u;
}

// ---------------------------------------------------------------------------
// column-wise sum / sum-of-squares over X[Nrows, C]   (C is 64 or 128, pow2)
// ---------------------------------------------------------------------------
__global__ void gat_colstats(const float* __restrict__ X, int Nrows, int C,
                             float* __restrict__ sums) {
  int col = threadIdx.x & (C - 1);
  int rpb = blockDim.x / C;
  int rl  = threadIdx.x / C;
  float s = 0.f, s2 = 0.f;
  for (int r = blockIdx.x * rpb + rl; r < Nrows; r += gridDim.x * rpb) {
    float v = X[(size_t)r * C + col];
    s += v; s2 += v * v;
  }
  atomicAdd(&sums[col], s);
  atomicAdd(&sums[C + col], s2);
}

// ss[0:C]=scale (gamma*rsqrt(var+eps)), ss[C:2C]=shift (beta - mu*scale)
__global__ void gat_bn_finalize(const float* __restrict__ sums,
                                const float* __restrict__ gamma,
                                const float* __restrict__ beta,
                                int Nrows, int C, float* __restrict__ ss) {
  int c = blockIdx.x * blockDim.x + threadIdx.x;
  if (c >= C) return;
  float inv = 1.0f / (float)Nrows;
  float mu  = sums[c] * inv;
  float var = fmaxf(sums[C + c] * inv - mu * mu, 0.0f);
  float sc  = gamma[c] * rsqrtf(var + BNEPS);
  ss[c]     = sc;
  ss[C + c] = beta[c] - mu * sc;
}

// ---------------------------------------------------------------------------
// WMMA GEMM: OUT[M, TN*16] = act( pre(A)[M,K] @ W[K, TN*16] + bias )
//
// W is staged once per block into LDS, converted to f16 and pre-swizzled into
// the exact per-lane B-fragment layout [kstep][tile][lane][16 halves], so each
// wave fetches a B fragment as one contiguous 32-byte LDS read.  A fragments
// are two 8-float contiguous runs per lane -> 4x global_load_b128 per K-step.
// One wave32 owns a 16-row band and all TN column tiles (A-frag reuse).
// ---------------------------------------------------------------------------
template <int TN>
__launch_bounds__(256)
__global__ void gat_wmma_gemm(const float* __restrict__ A,
                              const float* __restrict__ W,
                              const float* __restrict__ bias,   // may be null
                              const float* __restrict__ bnSS,   // may be null, [2K]
                              float* __restrict__ OUT,
                              int M, int K, int aRelu, int outRelu) {
  constexpr int Cout = TN * 16;
  __shared__ _Float16 Wlds[4 * TN * 32 * 16];   // up to K=128 (4 K-steps)

  // ---- cooperative fill: coalesced row-major read of W, swizzled f16 store
  {
    int total = K * Cout;
    for (int i = threadIdx.x; i < total; i += blockDim.x) {
      int k = i / Cout;            // Cout is constexpr pow2 -> shift
      int c = i - k * Cout;
      int kstep = k >> 5;
      int gg    = (k >> 3) & 1;
      int e     = (k & 7) | (((k >> 4) & 1) << 3);
      int t     = c >> 4;
      int nl    = c & 15;
      int lane  = (gg << 4) | nl;
      Wlds[(((kstep * TN + t) << 5) + lane) * 16 + e] = (_Float16)W[i];
    }
  }
  __syncthreads();

  int wave = (int)((blockIdx.x * (long long)blockDim.x + threadIdx.x) >> 5);
  int lane = threadIdx.x & 31;
  int tilesM = (M + 15) >> 4;
  if (wave >= tilesM) return;          // uniform per wave: EXEC stays all-ones

  int g  = lane >> 4;                  // half-wave group (0/1)
  int nl = lane & 15;
  int row  = (wave << 4) + nl;
  int arow = row < M ? row : M - 1;

  v8f vz = {};
  v8f acc[TN];
#pragma unroll
  for (int t = 0; t < TN; ++t) acc[t] = vz;

  const float* Ar = A + (size_t)arow * K;
  for (int k0 = 0; k0 < K; k0 += 32) {
    // ---- A fragment: runs [k0+8g, +8) and [k0+16+8g, +8) -> 4x b128 loads
    const float* Ap = Ar + k0 + (g << 3);
    float4 p0 = *(const float4*)(Ap + 0);
    float4 p1 = *(const float4*)(Ap + 4);
    float4 p2 = *(const float4*)(Ap + 16);
    float4 p3 = *(const float4*)(Ap + 20);
    float va[16] = {p0.x, p0.y, p0.z, p0.w, p1.x, p1.y, p1.z, p1.w,
                    p2.x, p2.y, p2.z, p2.w, p3.x, p3.y, p3.z, p3.w};
    if (bnSS) {
      const float* Sc = bnSS + k0 + (g << 3);
      const float* Sh = Sc + K;
      float4 s0 = *(const float4*)(Sc + 0),  s1 = *(const float4*)(Sc + 4);
      float4 s2 = *(const float4*)(Sc + 16), s3 = *(const float4*)(Sc + 20);
      float4 h0 = *(const float4*)(Sh + 0),  h1 = *(const float4*)(Sh + 4);
      float4 h2 = *(const float4*)(Sh + 16), h3 = *(const float4*)(Sh + 20);
      float vs[16] = {s0.x, s0.y, s0.z, s0.w, s1.x, s1.y, s1.z, s1.w,
                      s2.x, s2.y, s2.z, s2.w, s3.x, s3.y, s3.z, s3.w};
      float vh[16] = {h0.x, h0.y, h0.z, h0.w, h1.x, h1.y, h1.z, h1.w,
                      h2.x, h2.y, h2.z, h2.w, h3.x, h3.y, h3.z, h3.w};
#pragma unroll
      for (int e = 0; e < 16; ++e) va[e] = va[e] * vs[e] + vh[e];
    }
    if (aRelu) {
#pragma unroll
      for (int e = 0; e < 16; ++e) va[e] = fmaxf(va[e], 0.0f);
    }
    v16h a;
#pragma unroll
    for (int e = 0; e < 16; ++e) a[e] = (_Float16)va[e];

    // ---- B fragments straight from LDS (32B contiguous per lane)
    int kstep = k0 >> 5;
#pragma unroll
    for (int t = 0; t < TN; ++t) {
      v16h b = *(const v16h*)&Wlds[(((kstep * TN + t) << 5) + lane) * 16];
      acc[t] = __builtin_amdgcn_wmma_f32_16x16x32_f16(
          false, a, false, b, (short)0, acc[t], false, false);
    }
  }

#pragma unroll
  for (int t = 0; t < TN; ++t) {
    int col = (t << 4) + nl;
    float bv = bias ? bias[col] : 0.0f;
#pragma unroll
    for (int v = 0; v < 8; ++v) {
      int r = (wave << 4) + (g << 3) + v;   // D layout: M = vgpr + 8*group
      if (r < M) {
        float val = acc[t][v] + bv;
        if (outRelu) val = fmaxf(val, 0.0f);
        OUT[(size_t)r * Cout + col] = val;
      }
    }
  }
}

// ---------------------------------------------------------------------------
// attention score dots: scs[n,h] = <hfeat[n, h*C:(h+1)*C], a_s[h]>, same for scd
// ---------------------------------------------------------------------------
__global__ void gat_scores(const float* __restrict__ hfeat,
                           const float* __restrict__ as_,
                           const float* __restrict__ ad_,
                           float* __restrict__ scs, float* __restrict__ scd,
                           int N, int H) {
  int tid = blockIdx.x * blockDim.x + threadIdx.x;
  if (tid >= N * H) return;
  int h = tid % H, n = tid / H;
  int C = 128 / H;
  const float* hp = hfeat + (size_t)n * 128 + h * C;
  const float* ap = as_ + h * C;
  const float* dp = ad_ + h * C;
  float ss = 0.f, sd = 0.f;
  for (int c = 0; c < C; c += 4) {
    float4 v = *(const float4*)(hp + c);
    float4 a = *(const float4*)(ap + c);
    float4 d = *(const float4*)(dp + c);
    ss += v.x * a.x + v.y * a.y + v.z * a.z + v.w * a.w;
    sd += v.x * d.x + v.y * d.y + v.z * d.z + v.w * d.w;
  }
  scs[tid] = ss; scd[tid] = sd;
}

// ---------------------------------------------------------------------------
// edge softmax pass 1: per-dst segment max (encoded atomicMax)
// edge id e in [0,E) = real edge; [E, E+N) = self loop (src=dst=e-E)
// ---------------------------------------------------------------------------
__global__ void gat_edge_max(const int* __restrict__ src, const int* __restrict__ dst,
                             int E, int N, int H,
                             const float* __restrict__ scs, const float* __restrict__ scd,
                             unsigned* __restrict__ mEnc) {
  int e = blockIdx.x * blockDim.x + threadIdx.x;
  if (e >= E + N) return;
  int s, d;
  if (e < E) { s = src[e]; d = dst[e]; } else { s = d = e - E; }
  for (int h = 0; h < H; ++h) {
    float v = scs[s * H + h] + scd[d * H + h];
    v = v >= 0.f ? v : NEGSLOPE * v;
    atomicMax(&mEnc[d * H + h], fenc(v));
  }
}

// pass 2: e_exp = exp(val - max[dst]); store per-edge, atomicAdd per-dst sum
__global__ void gat_edge_exp(const int* __restrict__ src, const int* __restrict__ dst,
                             int E, int N, int H,
                             const float* __restrict__ scs, const float* __restrict__ scd,
                             const unsigned* __restrict__ mEnc,
                             float* __restrict__ eExp, float* __restrict__ ssum) {
  int e = blockIdx.x * blockDim.x + threadIdx.x;
  if (e >= E + N) return;
  int s, d;
  if (e < E) { s = src[e]; d = dst[e]; } else { s = d = e - E; }
  for (int h = 0; h < H; ++h) {
    float v = scs[s * H + h] + scd[d * H + h];
    v = v >= 0.f ? v : NEGSLOPE * v;
    float m  = fdec(mEnc[d * H + h]);
    float ee = __expf(v - m);
    eExp[(size_t)e * H + h] = ee;
    atomicAdd(&ssum[d * H + h], ee);
  }
}

// pass 3: out[dst] += alpha * hfeat[src]; 32 threads/edge, float4 per thread
__global__ void gat_edge_scatter(const int* __restrict__ src, const int* __restrict__ dst,
                                 int E, int N, int H,
                                 const float* __restrict__ hfeat,
                                 const float* __restrict__ eExp,
                                 const float* __restrict__ ssum,
                                 float* __restrict__ out) {
  long long tid = (long long)blockIdx.x * blockDim.x + threadIdx.x;
  long long total = (long long)(E + N) * 32;
  if (tid >= total) return;
  int e  = (int)(tid >> 5);
  int c0 = ((int)tid & 31) << 2;       // channel 0..124, step 4
  int s, d;
  if (e < E) { s = src[e]; d = dst[e]; } else { s = d = e - E; }
  int Cper = 128 / H;
  int h = c0 / Cper;
  float alpha = eExp[(size_t)e * H + h] / (ssum[(size_t)d * H + h] + 1e-16f);
  const float4 hv = *(const float4*)(hfeat + (size_t)s * 128 + c0);
  float* op = out + (size_t)d * 128 + c0;
  atomicAdd(op + 0, alpha * hv.x);
  atomicAdd(op + 1, alpha * hv.y);
  atomicAdd(op + 2, alpha * hv.z);
  atomicAdd(op + 3, alpha * hv.w);
}

// ---------------------------------------------------------------------------
// final residual: hfin = relu(bn3(g) + h_skip)      (C = 128)
// ---------------------------------------------------------------------------
__global__ void gat_bn_skip_relu(const float* __restrict__ g,
                                 const float* __restrict__ ss,
                                 const float* __restrict__ h_skip,
                                 float* __restrict__ out, int N) {
  long long i = (long long)blockIdx.x * blockDim.x + threadIdx.x;
  if (i >= (long long)N * 128) return;
  int c = (int)(i & 127);
  out[i] = fmaxf(g[i] * ss[c] + ss[128 + c] + h_skip[i], 0.0f);
}

// ---------------------------------------------------------------------------
// head: logits = c2 @ W3 + b3 (32->2), conf = sigmoid(hfin @ confW + confb)
// output layout: [N*2 logits][N conf]
// ---------------------------------------------------------------------------
__global__ void gat_head(const float* __restrict__ hfin, const float* __restrict__ c2,
                         const float* __restrict__ W3, const float* __restrict__ b3,
                         const float* __restrict__ confW, const float* __restrict__ confb,
                         float* __restrict__ outp, int N) {
  int n = blockIdx.x * blockDim.x + threadIdx.x;
  if (n >= N) return;
  float l0 = b3[0], l1 = b3[1];
  const float* cp = c2 + (size_t)n * 32;
#pragma unroll
  for (int c = 0; c < 32; ++c) { float v = cp[c]; l0 += v * W3[c * 2]; l1 += v * W3[c * 2 + 1]; }
  float ca = confb[0];
  const float* hp = hfin + (size_t)n * 128;
#pragma unroll
  for (int c = 0; c < 128; c += 4) {
    float4 v = *(const float4*)(hp + c);
    float4 w = *(const float4*)(confW + c);
    ca += v.x * w.x + v.y * w.y + v.z * w.z + v.w * w.w;
  }
  outp[(size_t)n * 2 + 0] = l0;
  outp[(size_t)n * 2 + 1] = l1;
  outp[(size_t)2 * N + n] = 1.0f / (1.0f + __expf(-ca));
}

// ===========================================================================
extern "C" void kernel_launch(void* const* d_in, const int* in_sizes, int n_in,
                              void* d_out, int out_size, void* d_ws, size_t ws_size,
                              hipStream_t stream) {
  const int N = in_sizes[0] / 64;
  const int E = in_sizes[1] / 2;

  const float* x        = (const float*)d_in[0];
  const int*   ei       = (const int*)d_in[1];
  const float* in_gamma = (const float*)d_in[2];
  const float* in_beta  = (const float*)d_in[3];
  const float* proj_W   = (const float*)d_in[4];
  const float* proj_b   = (const float*)d_in[5];
  const float* skip_W   = (const float*)d_in[6];
  const float* skip_b   = (const float*)d_in[7];
  const float* convW[3] = {(const float*)d_in[8],  (const float*)d_in[14], (const float*)d_in[20]};
  const float* convAs[3]= {(const float*)d_in[9],  (const float*)d_in[15], (const float*)d_in[21]};
  const float* convAd[3]= {(const float*)d_in[10], (const float*)d_in[16], (const float*)d_in[22]};
  const float* bnG[3]   = {(const float*)d_in[12], (const float*)d_in[18], (const float*)d_in[24]};
  const float* bnB[3]   = {(const float*)d_in[13], (const float*)d_in[19], (const float*)d_in[25]};
  const float* cls_W1   = (const float*)d_in[26];
  const float* cls_b1   = (const float*)d_in[27];
  const float* cls_W2   = (const float*)d_in[28];
  const float* cls_b2   = (const float*)d_in[29];
  const float* cls_W3   = (const float*)d_in[30];
  const float* cls_b3   = (const float*)d_in[31];
  const float* conf_W   = (const float*)d_in[32];
  const float* conf_b   = (const float*)d_in[33];
  // NOTE: conv biases (d_in[11],[17],[23]) are mathematically no-ops: each is
  // immediately followed by BatchNorm over axis 0, and BN(g + c_per_col) == BN(g).

  const int* srcA = ei;
  const int* dstA = ei + E;

  float* ws = (float*)d_ws;
  size_t off = 0;
  auto alloc = [&](size_t n) { float* p = ws + off; off += n; return p; };
  float* hr     = alloc((size_t)N * 128);        // also reused as hfin
  float* h_skip = alloc((size_t)N * 128);
  float* hfeat  = alloc((size_t)N * 128);
  float* bufA   = alloc((size_t)N * 128);
  float* bufB   = alloc((size_t)N * 128);        // later reused: c1 (N*64) + c2 (N*32)
  float* scs    = alloc((size_t)N * 4);
  float* scd    = alloc((size_t)N * 4);
  unsigned* mEnc= (unsigned*)alloc((size_t)N * 4);
  float* ssum   = alloc((size_t)N * 4);          // contiguous after mEnc
  float* eExp   = alloc((size_t)(E + N) * 4);
  float* stats  = alloc(256);
  float* ssin   = alloc(128);
  float* ssbn[3]= {alloc(256), alloc(256), alloc(256)};
  float* c1 = bufB;
  float* c2 = bufB + (size_t)N * 64;

  auto zero = [&](void* p, long long n) {
    int blk = (int)((n + 255) / 256);
    gat_zero<<<blk, 256, 0, stream>>>((unsigned*)p, n);
  };
  auto gemm = [&](const float* A, const float* W, const float* bias, const float* bnSS,
                  float* OUT, int K, int Cout, int aRelu, int outRelu) {
    int tilesM = (N + 15) / 16;
    int blocks = (tilesM + 7) / 8;   // 8 waves (wave32) per 256-thread block
    if (Cout == 128)
      gat_wmma_gemm<8><<<blocks, 256, 0, stream>>>(A, W, bias, bnSS, OUT, N, K, aRelu, outRelu);
    else if (Cout == 64)
      gat_wmma_gemm<4><<<blocks, 256, 0, stream>>>(A, W, bias, bnSS, OUT, N, K, aRelu, outRelu);
    else
      gat_wmma_gemm<2><<<blocks, 256, 0, stream>>>(A, W, bias, bnSS, OUT, N, K, aRelu, outRelu);
  };
  auto colbn = [&](const float* X, int C, const float* g, const float* b, float* ss) {
    zero(stats, 2 * C);
    gat_colstats<<<256, 256, 0, stream>>>(X, N, C, stats);
    gat_bn_finalize<<<(C + 127) / 128, 128, 0, stream>>>(stats, g, b, N, C, ss);
  };

  const int EB = ((E + N) + 255) / 256;
  auto gat_layer = [&](const float* Ain, const float* preSS, int preRelu, int layer,
                       int H, float* gout) {
    gemm(Ain, convW[layer], nullptr, preSS, hfeat, 128, 128, preRelu, 0);
    gat_scores<<<(N * H + 255) / 256, 256, 0, stream>>>(hfeat, convAs[layer], convAd[layer],
                                                        scs, scd, N, H);
    zero(mEnc, (long long)N * 8);            // mEnc + ssum (contiguous)
    zero(gout, (long long)N * 128);
    gat_edge_max<<<EB, 256, 0, stream>>>(srcA, dstA, E, N, H, scs, scd, mEnc);
    gat_edge_exp<<<EB, 256, 0, stream>>>(srcA, dstA, E, N, H, scs, scd, mEnc, eExp, ssum);
    long long st = (long long)(E + N) * 32;
    gat_edge_scatter<<<(int)((st + 255) / 256), 256, 0, stream>>>(srcA, dstA, E, N, H,
                                                                  hfeat, eExp, ssum, gout);
    colbn(gout, 128, bnG[layer], bnB[layer], ssbn[layer]);
  };

  // 1) input BN (folded into proj GEMM's A-load)
  colbn(x, 64, in_gamma, in_beta, ssin);
  // 2) hr = relu(bn(x) @ proj_W + proj_b)
  gemm(x, proj_W, proj_b, ssin, hr, 64, 128, 0, 1);
  // 3) h_skip = hr @ skip_W + skip_b
  gemm(hr, skip_W, skip_b, nullptr, h_skip, 128, 128, 0, 0);
  // 4-6) GAT layers (BN+ReLU of previous layer fused into each GEMM A-load)
  gat_layer(hr,   nullptr, 0, 0, 4, bufA);
  gat_layer(bufA, ssbn[0], 1, 1, 4, bufB);
  gat_layer(bufB, ssbn[1], 1, 2, 1, bufA);
  // 7) hfin = relu(bn3(g3) + h_skip)   (reuse hr)
  {
    long long n = (long long)N * 128;
    gat_bn_skip_relu<<<(int)((n + 255) / 256), 256, 0, stream>>>(bufA, ssbn[2], h_skip, hr, N);
  }
  // 8-9) classifier MLP via WMMA
  gemm(hr, cls_W1, cls_b1, nullptr, c1, 128, 64, 0, 1);
  gemm(c1, cls_W2, cls_b2, nullptr, c2, 64, 32, 0, 1);
  // 10) logits + confidence head
  gat_head<<<(N + 255) / 256, 256, 0, stream>>>(hr, c2, cls_W3, cls_b3, conf_W, conf_b,
                                                (float*)d_out, N);
  (void)n_in; (void)out_size; (void)ws_size;
}
